// ConvCapsuleLayer_86509231276877
// MI455X (gfx1250) — matrix operand order repaired and checked by reading, not compiled, not run on hardware
//
#include <hip/hip_runtime.h>
#include <hip/hip_bf16.h>

// ---- problem constants (from reference) ----
#define KK   5     // conv kernel size
#define ICN  4     // input capsules
#define IAN  16    // input atoms
#define NCN  8     // output capsules
#define NAN_ 16    // output atoms
#define OCN  (NCN*NAN_)  // 128 conv output channels
#define HH   128
#define WW   128
#define BBN  4
#define TM   16    // pixels per tile (GEMM M)
#define NTAP (KK*KK)     // 25
#define NCHUNK 13        // ceil(25/2) K-chunks of 2 taps (K=32)
#define ROUTE_ITERS 3

typedef __attribute__((ext_vector_type(16))) _Float16 v16h;
typedef __attribute__((ext_vector_type(8)))  float    v8f;

union AFrag { v16h v; _Float16 h[16]; float4 q[2]; };
union CFrag { v8f  v; float f[8]; };

__global__ __launch_bounds__(256)
void capsconv_route_kernel(const float* __restrict__ xin,
                           const float* __restrict__ Wp,
                           const float* __restrict__ bp,
                           float* __restrict__ out)
{
    // LDS: f16 input halo tile per input capsule + routing logits exchange
    __shared__ __align__(16) _Float16 inTile[ICN][KK][TM + 4][IAN]; // 12.8 KB
    __shared__ float logitsLDS[ICN][TM][NCN];                       // 2 KB

    const int tid  = threadIdx.x;
    const int lane = tid & 31;
    const int wcap = tid >> 5;          // wave id == output capsule j (8 waves)
    const int tilesX = WW / TM;         // 8
    const int tx = blockIdx.x % tilesX;
    const int y  = (blockIdx.x / tilesX) % HH;
    const int b  =  blockIdx.x / (tilesX * HH);
    const int x0 = tx * TM;

    // ---------------- stage input halo tile: fp32 -> f16 ----------------
    // x layout: [B][H][W][IC][IA]
    const int TOT = ICN * KK * (TM + 4) * IAN;  // 6400
    for (int i = tid; i < TOT; i += 256) {
        int ia  = i & (IAN - 1);
        int t   = i >> 4;
        int col = t % (TM + 4); t /= (TM + 4);
        int r   = t % KK;
        int ic  = t / KK;
        int gy = y + r - 2;
        int gx = x0 + col - 2;
        float v = 0.0f;
        if (gy >= 0 && gy < HH && gx >= 0 && gx < WW)
            v = xin[((((size_t)b * HH + gy) * WW + gx) * ICN + ic) * IAN + ia];
        inTile[ic][r][col][ia] = (_Float16)v;
    }
    __syncthreads();

    // ---------------- implicit-GEMM conv via WMMA ----------------
    const int m     = lane & 15;    // A row (pixel) / B col (atom)
    const int hi    = lane >> 4;    // half-wave selector
    const int iaOff = hi * 8;       // unused: each A lane carries all 16 atoms of one tap
    const int oc    = wcap * NAN_ + m;  // global conv output channel

    CFrag acc[ICN];                 // votes[ic][16px][16atoms] for capsule wcap
    #pragma unroll
    for (int ic = 0; ic < ICN; ++ic)
        #pragma unroll
        for (int e = 0; e < 8; ++e) acc[ic].f[e] = 0.0f;

    #pragma unroll
    for (int c = 0; c < NCHUNK; ++c) {
        // B fragment (32x16 f16): lanes 0-15 -> K 0..15 (tap 2c), 16-31 -> K 16..31 (tap 2c+1)
        AFrag bf;
        const int tB = 2 * c + hi;
        if (tB < NTAP) {
            #pragma unroll
            for (int ia = 0; ia < IAN; ++ia)   // coalesced: 16 consecutive oc per half-wave
                bf.h[ia] = (_Float16)Wp[(tB * IAN + ia) * OCN + oc];
        } else {
            #pragma unroll
            for (int ia = 0; ia < IAN; ++ia) bf.h[ia] = (_Float16)0.0f;
        }

        const int t0 = 2 * c, t1 = 2 * c + 1;
        const int ky0 = t0 / KK, kx0 = t0 % KK;
        const int ky1 = t1 / KK, kx1 = t1 % KK;

        #pragma unroll
        for (int ic = 0; ic < ICN; ++ic) {
            // A fragment (16x32 f16): lanes 0-15 -> tap 2c atoms 0..15, lanes 16-31 -> tap 2c+1 atoms 0..15
            AFrag af;
            if (tB < NTAP) {
                af.q[0] = *(const float4*)&inTile[ic][tB / KK][tB % KK + m][0];
                af.q[1] = *(const float4*)&inTile[ic][tB / KK][tB % KK + m][8];
            } else {
                af.q[0] = make_float4(0.f, 0.f, 0.f, 0.f);
                af.q[1] = make_float4(0.f, 0.f, 0.f, 0.f);
            }

            acc[ic].v = __builtin_amdgcn_wmma_f32_16x16x32_f16(
                false, af.v, false, bf.v, (short)0, acc[ic].v, false, false);
        }
    }

    // ---------------- dynamic routing (3 iters), votes stay in VGPRs ----------------
    // D layout: lane -> atom = m; element v -> pixel = v + 8*hi
    const float biasv = bp[wcap * NAN_ + m];
    float logits_l[ICN][8];
    #pragma unroll
    for (int ic = 0; ic < ICN; ++ic)
        #pragma unroll
        for (int v = 0; v < 8; ++v) logits_l[ic][v] = 0.0f;

    float actOut[8];

    for (int iter = 0; iter < ROUTE_ITERS; ++iter) {
        // publish per-(ic,pixel,capsule) logits (atom-0 lanes only)
        if (m == 0) {
            #pragma unroll
            for (int ic = 0; ic < ICN; ++ic)
                #pragma unroll
                for (int v = 0; v < 8; ++v)
                    logitsLDS[ic][v + 8 * hi][wcap] = logits_l[ic][v];
        }
        __syncthreads();

        // softmax over output capsules
        float route[ICN][8];
        #pragma unroll
        for (int ic = 0; ic < ICN; ++ic) {
            #pragma unroll
            for (int v = 0; v < 8; ++v) {
                const int px = v + 8 * hi;
                float mx = -1e30f;
                #pragma unroll
                for (int jj = 0; jj < NCN; ++jj)
                    mx = fmaxf(mx, logitsLDS[ic][px][jj]);
                float s = 0.0f;
                #pragma unroll
                for (int jj = 0; jj < NCN; ++jj)
                    s += __expf(logitsLDS[ic][px][jj] - mx);
                route[ic][v] = __expf(logits_l[ic][v] - mx) / s;
            }
        }
        __syncthreads();   // reads done before next iteration's writes

        // preactivate + squash + agreement
        #pragma unroll
        for (int v = 0; v < 8; ++v) {
            float p = biasv;
            #pragma unroll
            for (int ic = 0; ic < ICN; ++ic) p += route[ic][v] * acc[ic].f[v];

            // ||.||^2 over 16 atoms = butterfly across the 16-lane half-group
            float s2 = p * p;
            s2 += __shfl_xor(s2, 1);
            s2 += __shfl_xor(s2, 2);
            s2 += __shfl_xor(s2, 4);
            s2 += __shfl_xor(s2, 8);
            const float a = p * sqrtf(s2) / (1.0f + s2);   // t * ||t|| / (1+||t||^2)
            actOut[v] = a;

            #pragma unroll
            for (int ic = 0; ic < ICN; ++ic) {
                float d = acc[ic].f[v] * a;
                d += __shfl_xor(d, 1);
                d += __shfl_xor(d, 2);
                d += __shfl_xor(d, 4);
                d += __shfl_xor(d, 8);
                logits_l[ic][v] += d;
            }
        }
    }

    // ---------------- store activation [B,H,W,NC,NA] ----------------
    #pragma unroll
    for (int v = 0; v < 8; ++v) {
        const int px = v + 8 * hi;
        out[((((size_t)b * HH + y) * WW + (x0 + px)) * NCN + wcap) * NAN_ + m] = actOut[v];
    }
}

extern "C" void kernel_launch(void* const* d_in, const int* in_sizes, int n_in,
                              void* d_out, int out_size, void* d_ws, size_t ws_size,
                              hipStream_t stream) {
    (void)in_sizes; (void)n_in; (void)d_ws; (void)ws_size; (void)out_size;
    const float* x = (const float*)d_in[0];   // [4,128,128,4,16]
    const float* W = (const float*)d_in[1];   // [5,5,16,128]
    const float* b = (const float*)d_in[2];   // [1,1,8,16]
    float* out = (float*)d_out;               // [4,128,128,8,16]

    dim3 grid(BBN * HH * (WW / TM));          // 4096 blocks
    dim3 block(256);                          // 8 waves: one per output capsule
    capsconv_route_kernel<<<grid, block, 0, stream>>>(x, W, b, out);
}
